// GCNActor_91233695301736
// MI455X (gfx1250) — matrix operand order repaired and checked by reading, not compiled, not run on hardware
//
#include <hip/hip_runtime.h>

typedef __attribute__((ext_vector_type(16))) _Float16 v16h;
typedef __attribute__((ext_vector_type(8)))  _Float16 v8h;
typedef __attribute__((ext_vector_type(4)))  _Float16 v4h;
typedef __attribute__((ext_vector_type(8)))  float    v8f;

#define LDSPAD 136  // padded stride (halves): 272B rows -> lanes hit distinct LDS banks
#define MT 4        // M tiles (of 16 rows) per GEMM block

// ---------------- small utility kernels ----------------

__global__ void k_f32_to_f16(const float* __restrict__ in, _Float16* __restrict__ out, int n) {
  int i = blockIdx.x * blockDim.x + threadIdx.x;
  if (i < n) out[i] = (_Float16)in[i];
}

// W[k][c] (kdim x cdim row-major, f32) -> Wt[c][k] f16
__global__ void k_transpose_f16(const float* __restrict__ W, _Float16* __restrict__ Wt,
                                int kdim, int cdim) {
  int i = blockIdx.x * blockDim.x + threadIdx.x;
  if (i >= kdim * cdim) return;
  int k = i / cdim, c = i % cdim;
  Wt[c * kdim + k] = (_Float16)W[i];
}

__global__ void k_init_deg_cnt(float* __restrict__ deg, int* __restrict__ cnt, int n) {
  int i = blockIdx.x * blockDim.x + threadIdx.x;
  if (i < n) { deg[i] = 1.0f; cnt[i] = 0; }  // self-loop contributes 1 to degree
}

__global__ void k_count(const int* __restrict__ dst, float* __restrict__ deg,
                        int* __restrict__ cnt, int E) {
  int e = blockIdx.x * blockDim.x + threadIdx.x;
  if (e < E) { int d = dst[e]; atomicAdd(&deg[d], 1.0f); atomicAdd(&cnt[d], 1); }
}

__global__ void k_rsqrt_inplace(float* __restrict__ deg, int n) {
  int i = blockIdx.x * blockDim.x + threadIdx.x;
  if (i < n) deg[i] = rsqrtf(deg[i]);  // deg >= 1 always (self-loop)
}

__global__ void k_zero_int(int* __restrict__ p, int n) {
  int i = blockIdx.x * blockDim.x + threadIdx.x;
  if (i < n) p[i] = 0;
}

// single-block exclusive scan over node counts -> row_ptr (n ~100k: trivial cost)
__global__ __launch_bounds__(1024)
void k_scan(const int* __restrict__ cnt, int* __restrict__ row_ptr, int n) {
  __shared__ int sd[1024];
  __shared__ int run;
  if (threadIdx.x == 0) run = 0;
  __syncthreads();
  for (int base = 0; base < n; base += 1024) {
    int i = base + (int)threadIdx.x;
    int v = (i < n) ? cnt[i] : 0;
    sd[threadIdx.x] = v;
    __syncthreads();
    for (int off = 1; off < 1024; off <<= 1) {
      int t = (threadIdx.x >= (unsigned)off) ? sd[threadIdx.x - off] : 0;
      __syncthreads();
      sd[threadIdx.x] += t;
      __syncthreads();
    }
    int incl = sd[threadIdx.x];
    int r = run;
    if (i < n) row_ptr[i + 1] = r + incl;
    __syncthreads();
    if (threadIdx.x == 1023) run = r + sd[1023];
    __syncthreads();
  }
  if (threadIdx.x == 0) row_ptr[0] = 0;
}

// CSR fill; also precompute per-edge symmetric norm dinv[src]*dinv[dst]
__global__ void k_fill(const int* __restrict__ src, const int* __restrict__ dst,
                       const int* __restrict__ row_ptr, int* __restrict__ fill,
                       const float* __restrict__ dinv, int* __restrict__ csr_src,
                       float* __restrict__ csr_w, int E) {
  int e = blockIdx.x * blockDim.x + threadIdx.x;
  if (e >= E) return;
  int s = src[e], d = dst[e];
  int pos = row_ptr[d] + atomicAdd(&fill[d], 1);
  csr_src[pos] = s;
  csr_w[pos]   = dinv[s] * dinv[d];
}

// ---------------- WMMA GEMM: [n x 128](f16) @ [128 x 128](f16, pre-transposed) -> f16 ----------------
// block = 256 thr = 8 waves; block owns MT*16 rows, wave w owns cols [16w,16w+16).
// B fragments hoisted to registers and reused across the MT row-tiles.
__global__ __launch_bounds__(256)
void k_gemm_wmma(const _Float16* __restrict__ A, const _Float16* __restrict__ Bt,
                 _Float16* __restrict__ C, int nTiles) {
  __shared__ _Float16 sA[(16 * MT) * LDSPAD];
  __shared__ _Float16 sB[128 * LDSPAD];
  const int tid  = threadIdx.x;
  const int wave = tid >> 5, lane = tid & 31;
  const int tile0 = blockIdx.x * MT;

  // cooperative load: MT A tiles (16x128 halves each) -> sA, padded rows
#pragma unroll
  for (int t = 0; t < MT; ++t) {
    int mTile = tile0 + t;
    if (mTile < nTiles) {
      int i = tid * 8;                 // 2048 halves per tile
      int r = i >> 7, c = i & 127;
      *reinterpret_cast<v8h*>(&sA[(t * 16 + r) * LDSPAD + c]) =
          *reinterpret_cast<const v8h*>(&A[((size_t)mTile * 16 + r) * 128 + c]);
    }
  }
  // cooperative load: Bt 128x128 halves -> sB, padded rows (row = output col)
  for (int i = tid * 8; i < 128 * 128; i += 256 * 8) {
    int r = i >> 7, c = i & 127;
    *reinterpret_cast<v8h*>(&sB[r * LDSPAD + c]) =
        *reinterpret_cast<const v8h*>(&Bt[i]);
  }
  __syncthreads();

  const int mrow = lane & 15;
  const int hi   = lane >> 4;        // 0: low K half, 1: high K half (contiguous k = 16*hi + s, same in A and B)
  const int colBase = wave * 16;
  const int col  = colBase + mrow;

  // hoist B fragments (reused for every M tile)
  v16h bfrag[4];
#pragma unroll
  for (int kk = 0; kk < 4; ++kk) {
    const _Float16* bp = &sB[col * LDSPAD + kk * 32 + hi * 16];
    v8h blo = *reinterpret_cast<const v8h*>(bp);
    v8h bhi = *reinterpret_cast<const v8h*>(bp + 8);
    bfrag[kk] = __builtin_shufflevector(blo, bhi, 0,1,2,3,4,5,6,7,8,9,10,11,12,13,14,15);
  }

  const int rbase = (lane < 16) ? 0 : 8;
#pragma unroll
  for (int t = 0; t < MT; ++t) {
    int mTile = tile0 + t;
    if (mTile >= nTiles) break;      // wave-uniform
    v8f acc = {0.f, 0.f, 0.f, 0.f, 0.f, 0.f, 0.f, 0.f};
#pragma unroll
    for (int kk = 0; kk < 4; ++kk) {
      const _Float16* ap = &sA[(t * 16 + mrow) * LDSPAD + kk * 32 + hi * 16];
      v8h alo = *reinterpret_cast<const v8h*>(ap);
      v8h ahi = *reinterpret_cast<const v8h*>(ap + 8);
      v16h a = __builtin_shufflevector(alo, ahi, 0,1,2,3,4,5,6,7,8,9,10,11,12,13,14,15);
      acc = __builtin_amdgcn_wmma_f32_16x16x32_f16(false, a, false, bfrag[kk], (short)0,
                                                   acc, false, false);
    }
    _Float16* Cp = C + (size_t)mTile * 16 * 128 + colBase;
#pragma unroll
    for (int r = 0; r < 8; ++r)
      Cp[(size_t)(r + rbase) * 128 + mrow] = (_Float16)acc[r];
  }
}

// ---------------- aggregation: one wave per destination node ----------------
// out[v] = relu( sum_e hwh[csr_src[e]] * csr_w[e] + hwh[v]*dinv[v]^2 + bias )
// lane owns 4 contiguous f16 columns -> per-neighbor read = coalesced 256B wave read
__global__ __launch_bounds__(256)
void k_aggregate(const _Float16* __restrict__ hwh, const int* __restrict__ row_ptr,
                 const int* __restrict__ csr_src, const float* __restrict__ csr_w,
                 const float* __restrict__ dinv, const float* __restrict__ bias,
                 _Float16* __restrict__ outh, int n) {
  const int wave = threadIdx.x >> 5, lane = threadIdx.x & 31;
  const int v = blockIdx.x * 8 + wave;
  if (v >= n) return;                // wave-uniform
  const float dv = dinv[v];
  const int c0 = lane * 4;

  const v4h sv = *reinterpret_cast<const v4h*>(hwh + (size_t)v * 128 + c0);
  const float sw = dv * dv;
  float a0 = (float)sv[0] * sw, a1 = (float)sv[1] * sw;
  float a2 = (float)sv[2] * sw, a3 = (float)sv[3] * sw;

  const int e0 = row_ptr[v], e1 = row_ptr[v + 1];
  for (int e = e0; e < e1; ++e) {
    int s   = csr_src[e];            // wave-uniform
    float w = csr_w[e];
    if (e + 2 < e1)                  // prefetch a future neighbor row (L2-resident)
      __builtin_prefetch(hwh + (size_t)csr_src[e + 2] * 128, 0, 0);
    const v4h r = *reinterpret_cast<const v4h*>(hwh + (size_t)s * 128 + c0);
    a0 += (float)r[0] * w; a1 += (float)r[1] * w;
    a2 += (float)r[2] * w; a3 += (float)r[3] * w;
  }

  const float4 bv = *reinterpret_cast<const float4*>(bias + c0);
  a0 = fmaxf(a0 + bv.x, 0.f);
  a1 = fmaxf(a1 + bv.y, 0.f);
  a2 = fmaxf(a2 + bv.z, 0.f);
  a3 = fmaxf(a3 + bv.w, 0.f);

  v4h ov;
  ov[0] = (_Float16)a0; ov[1] = (_Float16)a1;
  ov[2] = (_Float16)a2; ov[3] = (_Float16)a3;
  *reinterpret_cast<v4h*>(outh + (size_t)v * 128 + c0) = ov;
}

// ---------------- heads: [n x 128] @ [128 x 16] x2 (mean, logstd) via WMMA ----------------
__global__ __launch_bounds__(256)
void k_heads_wmma(const _Float16* __restrict__ H, const _Float16* __restrict__ Wmt,
                  const _Float16* __restrict__ Wst, const float* __restrict__ bm,
                  const float* __restrict__ bs, float* __restrict__ outMean,
                  float* __restrict__ outStd, int n) {
  const int wave = threadIdx.x >> 5, lane = threadIdx.x & 31;
  const int tile = blockIdx.x * 8 + wave;
  if (tile * 16 >= n) return;        // wave-uniform
  const int mrow = lane & 15;
  const int hi   = lane >> 4;

  v8f accM = {0.f, 0.f, 0.f, 0.f, 0.f, 0.f, 0.f, 0.f};
  v8f accS = {0.f, 0.f, 0.f, 0.f, 0.f, 0.f, 0.f, 0.f};
  const size_t aBase = (size_t)(tile * 16 + mrow) * 128;

#pragma unroll
  for (int k0 = 0; k0 < 128; k0 += 32) {
    const _Float16* ap = &H[aBase + k0 + hi * 16];
    v8h alo = *reinterpret_cast<const v8h*>(ap);
    v8h ahi = *reinterpret_cast<const v8h*>(ap + 8);
    v16h a = __builtin_shufflevector(alo, ahi, 0,1,2,3,4,5,6,7,8,9,10,11,12,13,14,15);

    const _Float16* bpm = &Wmt[mrow * 128 + k0 + hi * 16];
    v8h m0 = *reinterpret_cast<const v8h*>(bpm);
    v8h m1 = *reinterpret_cast<const v8h*>(bpm + 8);
    v16h bmv = __builtin_shufflevector(m0, m1, 0,1,2,3,4,5,6,7,8,9,10,11,12,13,14,15);

    const _Float16* bps = &Wst[mrow * 128 + k0 + hi * 16];
    v8h s0 = *reinterpret_cast<const v8h*>(bps);
    v8h s1 = *reinterpret_cast<const v8h*>(bps + 8);
    v16h bsv = __builtin_shufflevector(s0, s1, 0,1,2,3,4,5,6,7,8,9,10,11,12,13,14,15);

    accM = __builtin_amdgcn_wmma_f32_16x16x32_f16(false, a, false, bmv, (short)0, accM,
                                                  false, false);
    accS = __builtin_amdgcn_wmma_f32_16x16x32_f16(false, a, false, bsv, (short)0, accS,
                                                  false, false);
  }

  const int col = mrow;
  const int rbase = (lane < 16) ? 0 : 8;
  const float bmc = bm[col], bsc = bs[col];
#pragma unroll
  for (int r = 0; r < 8; ++r) {
    int row = tile * 16 + r + rbase;
    outMean[(size_t)row * 16 + col] = accM[r] + bmc;
    outStd [(size_t)row * 16 + col] = accS[r] + bsc;
  }
}

// ---------------- host launcher ----------------

extern "C" void kernel_launch(void* const* d_in, const int* in_sizes, int n_in,
                              void* d_out, int out_size, void* d_ws, size_t ws_size,
                              hipStream_t stream) {
  (void)n_in; (void)out_size; (void)ws_size;
  const float* x  = (const float*)d_in[0];
  const int*   ei = (const int*)d_in[1];
  const float* W1 = (const float*)d_in[2];
  const float* b1 = (const float*)d_in[3];
  const float* W2 = (const float*)d_in[4];
  const float* b2 = (const float*)d_in[5];
  const float* Wm = (const float*)d_in[6];
  const float* bm = (const float*)d_in[7];
  const float* Ws = (const float*)d_in[8];
  const float* bs = (const float*)d_in[9];

  const int N = in_sizes[0] / 128;
  const int E = in_sizes[1] / 2;
  const int* srcA = ei;
  const int* dstA = ei + E;

  // workspace carve (aligned 256B)
  char* base = (char*)d_ws;
  size_t off = 0;
  auto carve = [&](size_t bytes) -> void* {
    void* p = base + off;
    off = (off + bytes + 255) & ~(size_t)255;
    return p;
  };
  _Float16* xh   = (_Float16*)carve((size_t)N * 128 * 2);  // also reused as h2 (f16)
  _Float16* h1h  = (_Float16*)carve((size_t)N * 128 * 2);
  _Float16* hwh  = (_Float16*)carve((size_t)N * 128 * 2);  // per-layer GEMM output (pre-agg)
  _Float16* W1t  = (_Float16*)carve(128 * 128 * 2);
  _Float16* W2t  = (_Float16*)carve(128 * 128 * 2);
  _Float16* Wmt  = (_Float16*)carve(16 * 128 * 2);
  _Float16* Wst  = (_Float16*)carve(16 * 128 * 2);
  float*    dinv = (float*)carve((size_t)N * 4);
  int*      cnt  = (int*)carve((size_t)N * 4);      // reused as fill cursor
  int*      rowp = (int*)carve(((size_t)N + 1) * 4);
  int*      csrc = (int*)carve((size_t)E * 4);
  float*    csrw = (float*)carve((size_t)E * 4);

  const int T = 256;
  auto cdiv = [](int a, int b) { return (a + b - 1) / b; };

  // precision conversion + weight transposes
  k_f32_to_f16<<<cdiv(N * 128, T), T, 0, stream>>>(x, xh, N * 128);
  k_transpose_f16<<<cdiv(128 * 128, T), T, 0, stream>>>(W1, W1t, 128, 128);
  k_transpose_f16<<<cdiv(128 * 128, T), T, 0, stream>>>(W2, W2t, 128, 128);
  k_transpose_f16<<<cdiv(128 * 16, T), T, 0, stream>>>(Wm, Wmt, 128, 16);
  k_transpose_f16<<<cdiv(128 * 16, T), T, 0, stream>>>(Ws, Wst, 128, 16);

  // degree / norm / CSR (with precomputed per-edge weights)
  k_init_deg_cnt<<<cdiv(N, T), T, 0, stream>>>(dinv, cnt, N);
  k_count<<<cdiv(E, T), T, 0, stream>>>(dstA, dinv, cnt, E);
  k_rsqrt_inplace<<<cdiv(N, T), T, 0, stream>>>(dinv, N);
  k_scan<<<1, 1024, 0, stream>>>(cnt, rowp, N);
  k_zero_int<<<cdiv(N, T), T, 0, stream>>>(cnt, N);
  k_fill<<<cdiv(E, T), T, 0, stream>>>(srcA, dstA, rowp, cnt, dinv, csrc, csrw, E);

  const int nTiles = cdiv(N, 16);
  const int gemmBlocks = cdiv(nTiles, MT);

  // layer 1
  k_gemm_wmma<<<gemmBlocks, T, 0, stream>>>(xh, W1t, hwh, nTiles);
  k_aggregate<<<cdiv(N, 8), T, 0, stream>>>(hwh, rowp, csrc, csrw, dinv, b1, h1h, N);
  // layer 2 (reuse xh buffer for h2)
  k_gemm_wmma<<<gemmBlocks, T, 0, stream>>>(h1h, W2t, hwh, nTiles);
  k_aggregate<<<cdiv(N, 8), T, 0, stream>>>(hwh, rowp, csrc, csrw, dinv, b2, xh, N);
  // heads
  float* outMean = (float*)d_out;
  float* outStd  = outMean + (size_t)N * 16;
  k_heads_wmma<<<cdiv(nTiles, 8), T, 0, stream>>>(xh, Wmt, Wst, bm, bs, outMean, outStd, N);
}
